// QuantLinearAWQ_39084202394118
// MI455X (gfx1250) — matrix operand order repaired and checked by reading, not compiled, not run on hardware
//
#include <hip/hip_runtime.h>

// ---------------------------------------------------------------------------
// AWQ-style 4-bit dequant GEMM for MI455X (gfx1250, wave32, WMMA).
//
// Roofline: qweight (int32 codes) = 180 MB dominates; compute is 2.9 GFLOP.
// => single streaming pass over qweight, coalesced b32 loads that land
//    directly in the v_wmma_f32_16x16x32_f16 B-fragment layout, fused
//    packed-f16 dequant (magic-number 1024+q trick, no masking needed since
//    codes are in [0,16)), f32 accumulation. K split 8x for wave-level
//    parallelism, deterministic partial-sum reduction (no float atomics).
// ---------------------------------------------------------------------------

typedef __attribute__((ext_vector_type(16))) _Float16     v16h;
typedef __attribute__((ext_vector_type(2)))  _Float16     h2;
typedef __attribute__((ext_vector_type(8)))  float        v8f;
typedef __attribute__((ext_vector_type(8)))  unsigned int v8u;

#define TOKENS   32
#define K_DIM    4096
#define N_DIM    11008
#define KSPLIT   8
#define KB_TOTAL (K_DIM / 32)            // 128 K-steps of 32
#define KB_PER_SPLIT (KB_TOTAL / KSPLIT) // 16

// ---------------------------------------------------------------------------
// Kernel 1: pack x (fp32 [32,4096]) into f16 WMMA A-fragment layout.
// Fragment f = kb*2 + h (h = token half). 16-bit A 16x32 layout used here (contiguous K, same as B):
//   lane<16:  element e -> K = e             (M = h*16 + lane)
//   lane>=16: element e -> K = e + 16        (M = h*16 + lane-16)
// Stored [f][lane][8 x u32]: a GEMM lane fetches its fragment with one
// contiguous 32B load (2x global_load_b128), perfectly coalesced.
// ---------------------------------------------------------------------------
__global__ void awq_pack_x_kernel(const float* __restrict__ x,
                                  unsigned int* __restrict__ apack) {
    int u = blockIdx.x * blockDim.x + threadIdx.x;   // 128*2*32*8 = 65536 uints
    if (u >= KB_TOTAL * 2 * 32 * 8) return;
    int e2   = u & 7;          // f16-pair index within lane
    int lane = (u >> 3) & 31;
    int f    = u >> 8;         // fragment index
    int kb   = f >> 1;
    int h    = f & 1;
    int e0   = e2 * 2;         // even f16 element
    int k0   = kb * 32 + e0 + ((lane >> 4) << 4);
    int m    = h * 16 + (lane & 15);
    _Float16 h0 = (_Float16)x[m * K_DIM + k0];
    _Float16 h1 = (_Float16)x[m * K_DIM + k0 + 1];
    unsigned int lo = (unsigned int)__builtin_bit_cast(unsigned short, h0);
    unsigned int hi = (unsigned int)__builtin_bit_cast(unsigned short, h1);
    apack[u] = lo | (hi << 16);
}

// ---------------------------------------------------------------------------
// Kernel 2: fused dequant + WMMA GEMM. One wave owns a 16-column N tile and
// one K-split; grid = (N/128, KSPLIT), block = 256 threads = 8 waves.
// ---------------------------------------------------------------------------
__global__ void __launch_bounds__(256)
awq_wmma_kernel(const int*   __restrict__ qw,      // [K, N] 4-bit codes in i32
                const int*   __restrict__ qz,      // [K/128, N] zero codes
                const float* __restrict__ sc,      // [K/128, N] scales
                const v8u*   __restrict__ apack,   // packed A fragments (f16)
                float*       __restrict__ partial) // [KSPLIT, 32, N]
{
    const int lane  = threadIdx.x & 31;
    const int wave  = threadIdx.x >> 5;
    const int n0    = (blockIdx.x * 8 + wave) * 16;   // 16-col tile base
    const int n     = n0 + (lane & 15);               // this lane's column
    const int khalf = (lane >> 4) << 4;               // 0 or 16 (B layout)
    const int kb0   = blockIdx.y * KB_PER_SPLIT;

    v8f acc0 = {};   // tokens  0..15 x cols n0..n0+15
    v8f acc1 = {};   // tokens 16..31 x cols n0..n0+15

    for (int g4 = 0; g4 < KB_PER_SPLIT / 4; ++g4) {   // one quant group / iter
        const int kb_base = kb0 + g4 * 4;
        const int grp     = kb_base >> 2;             // (kb*32)/128
        // f16 magic: (code | 0x6400) as f16 == 1024 + code (exact; codes<16,
        // so no masking is needed anywhere).
        const unsigned zh  = (unsigned)qz[(size_t)grp * N_DIM + n] | 0x6400u;
        const h2 hz2       = __builtin_bit_cast(h2, zh | (zh << 16));
        const _Float16 sh  = (_Float16)sc[(size_t)grp * N_DIM + n];
        const h2 s2        = {sh, sh};

#pragma unroll
        for (int j = 0; j < 4; ++j) {
            const int kb = kb_base + j;

            // A fragments: one coalesced 32B load each (L2-resident).
            v16h a0 = __builtin_bit_cast(v16h, apack[(kb * 2 + 0) * 32 + lane]);
            v16h a1 = __builtin_bit_cast(v16h, apack[(kb * 2 + 1) * 32 + lane]);

            // B fragment: lane = column n, K = kb*32 + khalf + i.
            // Each load i: lanes 0-15 hit 64B contiguous of row k+i,
            // lanes 16-31 hit 64B contiguous of row k+16+i.
            const int* qcol = qw + (size_t)(kb * 32 + khalf) * N_DIM + n;
            unsigned q[16];
#pragma unroll
            for (int i = 0; i < 16; ++i)
                q[i] = (unsigned)qcol[(size_t)i * N_DIM]; // i*N*4 fits 24-bit imm

            union { v16h h; unsigned int u[8]; } b;
#pragma unroll
            for (int p = 0; p < 8; ++p) {
                // codes < 16 => upper bits already zero: lshl + or3 only
                unsigned packed = q[2 * p] | (q[2 * p + 1] << 16) | 0x64006400u;
                h2 hq   = __builtin_bit_cast(h2, packed);
                h2 diff = hq - hz2;                   // (q - z), exact in f16
                h2 bb   = diff * s2;                  // one rounding (pk_mul)
                b.u[p]  = __builtin_bit_cast(unsigned int, bb);
            }

            acc0 = __builtin_amdgcn_wmma_f32_16x16x32_f16(
                       false, a0, false, b.h, (short)0, acc0, false, false);
            acc1 = __builtin_amdgcn_wmma_f32_16x16x32_f16(
                       false, a1, false, b.h, (short)0, acc1, false, false);
        }
    }

    // C/D layout: VGPR v -> M=v (lanes 0-15), M=v+8 (lanes 16-31).
    float* outp = partial + (size_t)blockIdx.y * TOKENS * N_DIM;
    const int mrow = (lane >> 4) << 3;
#pragma unroll
    for (int v = 0; v < 8; ++v) {
        outp[(size_t)(mrow + v) * N_DIM + n]      = acc0[v];
        outp[(size_t)(16 + mrow + v) * N_DIM + n] = acc1[v];
    }
}

// ---------------------------------------------------------------------------
// Kernel 3: deterministic reduction of K-split partials + bias.
// ---------------------------------------------------------------------------
__global__ void awq_finalize_kernel(const float* __restrict__ partial,
                                    const float* __restrict__ bias,
                                    float* __restrict__ out) {
    int idx = blockIdx.x * blockDim.x + threadIdx.x;
    if (idx >= TOKENS * N_DIM) return;
    int nn = idx % N_DIM;
    float s = bias[nn];
#pragma unroll
    for (int p = 0; p < KSPLIT; ++p)
        s += partial[(size_t)p * TOKENS * N_DIM + idx];
    out[idx] = s;
}

extern "C" void kernel_launch(void* const* d_in, const int* in_sizes, int n_in,
                              void* d_out, int out_size, void* d_ws, size_t ws_size,
                              hipStream_t stream) {
    const float* x    = (const float*)d_in[0];
    const int*   qw   = (const int*)d_in[1];
    const int*   qz   = (const int*)d_in[2];
    const float* sc   = (const float*)d_in[3];
    const float* bias = (const float*)d_in[4];
    float*       out  = (float*)d_out;

    // Workspace layout: [0, 256KB) packed A fragments; then KSPLIT partials
    // (8 * 32 * 11008 * 4B ~= 11.3 MB).
    unsigned int* apack   = (unsigned int*)d_ws;
    float*        partial = (float*)((char*)d_ws + 256 * 1024);

    // 1) pack x -> f16 A fragments (65536 u32 outputs)
    awq_pack_x_kernel<<<(KB_TOTAL * 2 * 32 * 8) / 256, 256, 0, stream>>>(x, apack);

    // 2) streaming dequant-WMMA GEMM: grid (N/128 tiles, KSPLIT)
    awq_wmma_kernel<<<dim3(N_DIM / 128, KSPLIT), 256, 0, stream>>>(
        qw, qz, sc, (const v8u*)apack, partial);

    // 3) reduce partials + bias
    awq_finalize_kernel<<<(TOKENS * N_DIM + 255) / 256, 256, 0, stream>>>(
        partial, bias, out);
}
